// QLinearPerGrp_25031069401195
// MI455X (gfx1250) — matrix operand order repaired and checked by reading, not compiled, not run on hardware
//
#include <hip/hip_runtime.h>
#include <hip/hip_bf16.h>

// QLinear per-group (AWQ-style): y[8,11008] = x[8,4096] @ ((q - z) * s)^T
// HBM-bound on qweight (~180 MB int32 -> ~7.7us floor @ 23.3 TB/s).
// Strategy:
//   * stream qweight once, 64B contiguous per lane, v_wmma_f32_16x16x32_f16
//   * A (x as f16) fully preloaded in registers per wave (16 frags, 128 VGPRs)
//   * bit-trick dequant: f16(0x6400 | q) == 1024 + q exactly for q in [0,15]
//     -> B fragment built with 2 bitwise ops per dword, zero cvt instructions
//   * group-factored correction:
//       sum_k x*(q-z)*s = s*(sum_k x*(q+1024)) - s*(z+1024)*X[m,g]
//     where X[m,g] = sum_{k in g} f16(x[m,k])  (f16-rounded to cancel exactly)

#define M_TOK 8
#define K_IN  4096
#define N_OUT 11008
#define GSZ   128
#define NGRP  (K_IN / GSZ)      // 32
#define KPW   (K_IN / 8)        // 512 K-columns per wave (split-K over 8 waves)
#define GPW   (KPW / GSZ)       // 4 groups per wave
#define FPW   (KPW / 32)        // 16 A-fragments per wave

typedef __attribute__((ext_vector_type(16))) _Float16     v16h;
typedef __attribute__((ext_vector_type(8)))  float        v8f;
typedef __attribute__((ext_vector_type(8)))  unsigned int v8u;

__global__ __launch_bounds__(256)
void qlinear_wmma_kernel(const float* __restrict__ x,
                         const int*   __restrict__ qw,
                         const int*   __restrict__ qz,
                         const float* __restrict__ sc,
                         float*       __restrict__ out)
{
    __shared__ float xsum[16 * NGRP];        // X[m][g], rows 8..15 zero
    __shared__ float red[8 * 256];           // 8 waves x (16x16) partial tiles

    const int lane = threadIdx.x & 31;
    const int wave = threadIdx.x >> 5;       // 0..7, split-K id
    const int n0   = blockIdx.x * 16;        // N tile base (688 tiles)
    const int col  = lane & 15;              // N within tile; also A-frag row
    const int hi   = lane >> 4;              // half-wave K-phase select

    const int n     = n0 + col;              // weight row this lane handles
    const int kBase = wave * KPW;            // wave K range [kBase, kBase+512)

    // ---- X[m][g] = sum over group of f16-rounded x[m][k], once per block ----
    xsum[threadIdx.x]       = 0.0f;
    xsum[256 + threadIdx.x] = 0.0f;
    __syncthreads();
    {
        const int mm = threadIdx.x >> 5;     // 0..7
        const int gg = threadIdx.x & 31;     // 0..31
        const float* xp = x + mm * K_IN + gg * GSZ;
        float sacc = 0.0f;
        for (int i = 0; i < GSZ; i += 4) {
            sacc += (float)(_Float16)xp[i]     + (float)(_Float16)xp[i + 1]
                  + (float)(_Float16)xp[i + 2] + (float)(_Float16)xp[i + 3];
        }
        xsum[mm * NGRP + gg] = sacc;
    }
    __syncthreads();

    // ---- preload all 16 A fragments (x -> f16) into registers ----
    // lane holds M = col; half h in [0,16): K = kf + hi*16 + h
    v16h afrag[FPW];
    const int  m      = col;
    const bool mValid = (m < M_TOK);
    if (mValid) {
        const float* xr = x + m * K_IN + kBase + hi * 16;
#pragma unroll
        for (int f = 0; f < FPW; ++f) {
            const float* xp = xr + f * 32;
#pragma unroll
            for (int h = 0; h < 8; ++h) {
                afrag[f][h]     = (_Float16)xp[h];
                afrag[f][h + 8] = (_Float16)xp[8 + h];
            }
        }
    } else {
#pragma unroll
        for (int f = 0; f < FPW; ++f)
#pragma unroll
            for (int h = 0; h < 16; ++h) afrag[f][h] = (_Float16)0.0f;
    }

    v8f acc = {};

    for (int g = 0; g < GPW; ++g) {
        const int   grp = (kBase >> 7) + g;  // global group index
        const float s   = sc[n * NGRP + grp];
        const float zf  = (float)qz[n * NGRP + grp];

        // group-local accumulator: sum_k x*(q+1024) over this 128-K group
        v8f accg = {};
#pragma unroll
        for (int kk = 0; kk < GSZ / 32; ++kk) {
            // B fragment via bit pack: half = 0x6400 | q  ==  1024 + q (exact).
            // B 32x16, same K order as A: lane holds N = col; half h: K = k0 + hi*16 + h
            const int* qp = qw + n * K_IN + kBase + g * GSZ + kk * 32 + hi * 16;
            v8u braw;
#pragma unroll
            for (int j = 0; j < 8; ++j) {
                const unsigned q0 = (unsigned)qp[2 * j];
                const unsigned q1 = (unsigned)qp[2 * j + 1];
                braw[j] = 0x64006400u | (q1 << 16) | q0;
            }
            const v16h b = __builtin_bit_cast(v16h, braw);
            accg = __builtin_amdgcn_wmma_f32_16x16x32_f16(
                false, afrag[g * 4 + kk], false, b,
                (short)0, accg, false, false);
        }

        // fold: acc += s*accg - s*(z+1024)*X[row][g]
        const float sz = s * (zf + 1024.0f);
#pragma unroll
        for (int r = 0; r < 8; ++r) {
            const float xg = xsum[(r + hi * 8) * NGRP + grp];
            acc[r] = __builtin_fmaf(s, accg[r], acc[r]);
            acc[r] = __builtin_fmaf(-sz, xg, acc[r]);
        }
    }

    // ---- split-K reduction across the 8 waves via LDS ----
    // C/D layout: VGPR r, lane L -> row = r + (L/16)*8, col = L%16
#pragma unroll
    for (int r = 0; r < 8; ++r) {
        red[wave * 256 + (r + hi * 8) * 16 + col] = acc[r];
    }
    __syncthreads();

    if (threadIdx.x < 128) {                 // 8 rows x 16 cols of real output
        const int mm = threadIdx.x >> 4;
        const int nn = threadIdx.x & 15;
        float sum = 0.0f;
#pragma unroll
        for (int w = 0; w < 8; ++w) sum += red[w * 256 + mm * 16 + nn];
        out[mm * N_OUT + n0 + nn] = sum;
    }
}

extern "C" void kernel_launch(void* const* d_in, const int* in_sizes, int n_in,
                              void* d_out, int out_size, void* d_ws, size_t ws_size,
                              hipStream_t stream) {
    const float* x  = (const float*)d_in[0];   // [8, 4096]  f32
    const int*   qw = (const int*)  d_in[1];   // [11008, 4096] int32 in [0,15]
    const int*   qz = (const int*)  d_in[2];   // [11008, 32]   int32 in [0,15]
    const float* sc = (const float*)d_in[3];   // [11008, 32]   f32
    float*       out = (float*)d_out;          // [8, 11008] f32

    qlinear_wmma_kernel<<<N_OUT / 16, 256, 0, stream>>>(x, qw, qz, sc, out);
}
